// SpanParsingModel_65721589563998
// MI455X (gfx1250) — matrix operand order repaired and checked by reading, not compiled, not run on hardware
//
#include <hip/hip_runtime.h>

#define B_ 4
#define T_ 258
#define D_ 768
#define NW 128
#define HID 2048
#define NC 768
#define NP 256          // pieces = T-2
#define F1ROWS 130
#define GROWS 129       // per-batch G rows (j in [0,129))
#define MPAD 528        // 4*129=516 padded to 16-multiple
#define LN_EPS 1e-5f

typedef _Float16 half_t;
typedef _Float16 v16h __attribute__((ext_vector_type(16)));
typedef float    v8f  __attribute__((ext_vector_type(8)));
typedef float    v2f  __attribute__((ext_vector_type(2)));

// ---------------- Kernel 1: segment means -> f1 (4,130,768) ----------------
__global__ void k_f1(const float* __restrict__ f0, const int* __restrict__ seg,
                     float* __restrict__ f1) {
  int b = blockIdx.x / F1ROWS;
  int t = blockIdx.x % F1ROWS;
  int tid = threadIdx.x;                       // 256 threads
  float* dst = f1 + (b * F1ROWS + t) * D_;
  if (t == 0) {
    const float* src = f0 + (size_t)(b * T_) * D_;
    for (int c = tid; c < D_; c += 256) dst[c] = src[c];
    return;
  }
  if (t == F1ROWS - 1) {
    const float* src = f0 + (size_t)(b * T_ + T_ - 1) * D_;
    for (int c = tid; c < D_; c += 256) dst[c] = src[c];
    return;
  }
  int w = t - 1;
  __shared__ int ss[NP];
  ss[tid] = seg[b * NP + tid];
  __syncthreads();
  float a0 = 0.f, a1 = 0.f, a2 = 0.f;
  int cnt = 0;
  for (int p = 0; p < NP; ++p) {
    if (ss[p] == w) {
      const float* src = f0 + (size_t)(b * T_ + 1 + p) * D_;
      a0 += src[tid]; a1 += src[tid + 256]; a2 += src[tid + 512];
      ++cnt;
    }
  }
  float inv = 1.0f / (float)(cnt > 0 ? cnt : 1);
  dst[tid] = a0 * inv; dst[tid + 256] = a1 * inv; dst[tid + 512] = a2 * inv;
}

// ---------------- Kernel 2: gather G rows (528 x 768, zero-padded) ----------
__global__ void k_gather(const float* __restrict__ f1, float* __restrict__ G) {
  int r = blockIdx.x;                          // 0..527
  int tid = threadIdx.x;                       // 256
  float* dst = G + (size_t)r * D_;
  if (r >= B_ * GROWS) {
    dst[tid] = 0.f; dst[tid + 256] = 0.f; dst[tid + 512] = 0.f;
    return;
  }
  int b = r / GROWS, k = r % GROWS;
  const float* base = f1 + (size_t)b * F1ROWS * D_;
  for (int c = tid; c < D_; c += 256) {
    float v = (c < 384) ? base[k * D_ + 2 * c + 1]
                        : base[(k + 1) * D_ + 2 * (c - 384)];
    dst[c] = v;
  }
}

// ---------------- Kernel 3: w2 (768x2048 f32) -> f16 ------------------------
__global__ void k_w2h(const float* __restrict__ w2, half_t* __restrict__ w2h) {
  int i = blockIdx.x * blockDim.x + threadIdx.x;   // one float4 each
  float4 v = ((const float4*)w2)[i];
  union { half_t h[4]; uint2 u; } pk;
  pk.h[0] = (half_t)v.x; pk.h[1] = (half_t)v.y;
  pk.h[2] = (half_t)v.z; pk.h[3] = (half_t)v.w;
  ((uint2*)w2h)[i] = pk.u;
}

// ---------------- Kernel 4: HG = G @ w1^T  (528 x 2048), fp32 WMMA ----------
__global__ void k_hg(const float* __restrict__ G, const float* __restrict__ w1,
                     float* __restrict__ HG) {
  const int lane = threadIdx.x & 31, wave = threadIdx.x >> 5;
  const int g = lane >> 4, ln = lane & 15;
  int tile = blockIdx.x * 8 + wave;            // 528 blocks * 8 waves = 4224 tiles
  int tm = tile >> 7;                          // M tile 0..32
  int tn = tile & 127;                         // N tile 0..127
  int m0 = tm * 16, n0 = tn * 16;
  const float* ga = G  + (size_t)(m0 + ln) * D_;
  const float* gb = w1 + (size_t)(n0 + ln) * D_;
  v8f acc = {};
  // A 16x4 f32 layout: lane M=ln, VGPR0/1 = K = 2g, 2g+1  (contiguous b64)
  // B  4x16 f32 layout (mirrored): lane N=ln, K = 2g, 2g+1
#pragma unroll 4
  for (int k0 = 0; k0 < D_; k0 += 4) {
    v2f a = *(const v2f*)(ga + k0 + 2 * g);
    v2f b = *(const v2f*)(gb + k0 + 2 * g);
    acc = __builtin_amdgcn_wmma_f32_16x16x4_f32(false, a, false, b,
                                                (short)0, acc, false, false);
  }
  // D 16x16 f32: VGPR r -> (M = r + 8*g, N = ln)
  float* outp = HG + n0 + ln;
#pragma unroll
  for (int r = 0; r < 8; ++r)
    outp[(size_t)(m0 + r + 8 * g) * HID] = acc[r];
}

// ---------------- Kernel 5: fused h-build + LN + ReLU + f16 WMMA out-proj ---
// 1032 blocks x 512 threads (16 waves), 256KB dynamic LDS (64 rows x 2048 f16)
__global__ void __launch_bounds__(512) k_main(
    const float* __restrict__ HG, const float* __restrict__ b1,
    const float* __restrict__ gamma, const float* __restrict__ beta,
    const half_t* __restrict__ w2h, const float* __restrict__ b2,
    float* __restrict__ out) {
  extern __shared__ __align__(16) char smem[];   // 64 * 4096 bytes
  const int tid = threadIdx.x;
  const int lane = tid & 31, wave = tid >> 5;    // 16 waves
  const int g = lane >> 4, ln = lane & 15;
  const int bm = blockIdx.x;                     // rows [bm*64, bm*64+64)

  // ---- phase 1: normalized f16 rows into LDS (4 rows per wave) ----
  for (int rr = 0; rr < 4; ++rr) {
    int r = wave * 4 + rr;
    int R = bm * 64 + r;
    int b = R / (NW * GROWS);
    int rem = R % (NW * GROWS);
    int i = rem / GROWS, j = rem % GROWS;
    float vmask = (j > i) ? 1.0f : 0.0f;
    const float* hj = HG + (size_t)(b * GROWS + j) * HID;
    const float* hi = HG + (size_t)(b * GROWS + i) * HID;
    float4 h4[16];
    float s1 = 0.f, s2 = 0.f;
#pragma unroll
    for (int q = 0; q < 16; ++q) {
      int k = q * 128 + lane * 4;
      float4 aj = *(const float4*)(hj + k);
      float4 ai = *(const float4*)(hi + k);
      float4 bb = *(const float4*)(b1 + k);
      float4 h;
      h.x = vmask * (aj.x - ai.x) + bb.x;
      h.y = vmask * (aj.y - ai.y) + bb.y;
      h.z = vmask * (aj.z - ai.z) + bb.z;
      h.w = vmask * (aj.w - ai.w) + bb.w;
      h4[q] = h;
      s1 += h.x + h.y + h.z + h.w;
      s2 += h.x * h.x + h.y * h.y + h.z * h.z + h.w * h.w;
    }
#pragma unroll
    for (int off = 16; off > 0; off >>= 1) {
      s1 += __shfl_xor(s1, off, 32);
      s2 += __shfl_xor(s2, off, 32);
    }
    float mu = s1 * (1.0f / HID);
    float var = s2 * (1.0f / HID) - mu * mu;
    float rinv = rsqrtf(var + LN_EPS);
#pragma unroll
    for (int q = 0; q < 16; ++q) {
      int k = q * 128 + lane * 4;
      float4 gg = *(const float4*)(gamma + k);
      float4 be = *(const float4*)(beta + k);
      float4 h = h4[q];
      float y0 = fmaxf((h.x - mu) * rinv * gg.x + be.x, 0.f);
      float y1 = fmaxf((h.y - mu) * rinv * gg.y + be.y, 0.f);
      float y2 = fmaxf((h.z - mu) * rinv * gg.z + be.z, 0.f);
      float y3 = fmaxf((h.w - mu) * rinv * gg.w + be.w, 0.f);
      union { half_t f[4]; uint2 u; } pk;
      pk.f[0] = (half_t)y0; pk.f[1] = (half_t)y1;
      pk.f[2] = (half_t)y2; pk.f[3] = (half_t)y3;
      *(uint2*)(smem + r * 4096 + k * 2) = pk.u;
    }
  }
  __syncthreads();

  // ---- phase 2: [64x2048]f16 (LDS) x w2^T [2048x768]f16 -> fp32 out ----
  const int wm = wave & 3;                 // M tile (rows wm*16..+16)
  const int nbase = (wave >> 2) * 12;      // 12 N tiles per wave
  v8f acc[12];
#pragma unroll
  for (int t = 0; t < 12; ++t) acc[t] = (v8f){0, 0, 0, 0, 0, 0, 0, 0};

  const int arow = wm * 16 + ln;
  for (int kb = 0; kb < HID; kb += 32) {
    // A 16x32 f16 layout: lane M=ln; K in [16g, 16g+16) -> 32 contiguous bytes
    union { uint4 u[2]; v16h v; } a;
    const char* abase = smem + arow * 4096 + kb * 2 + g * 32;
    a.u[0] = *(const uint4*)(abase);
    a.u[1] = *(const uint4*)(abase + 16);
#pragma unroll
    for (int t = 0; t < 12; ++t) {
      int c = (nbase + t) * 16 + ln;
      // B 32x16 f16 layout: lane N=ln; K in [16g, 16g+16) -> 32 contiguous bytes
      union { uint4 u[2]; v16h v; } bm_;
      const char* bbase = (const char*)w2h + (size_t)c * (HID * 2) + (kb + 16 * g) * 2;
      bm_.u[0] = *(const uint4*)(bbase);
      bm_.u[1] = *(const uint4*)(bbase + 16);
      acc[t] = __builtin_amdgcn_wmma_f32_16x16x32_f16(false, a.v, false, bm_.v,
                                                      (short)0, acc[t], false, false);
    }
  }

  // epilogue: D VGPR r -> (M = r + 8g, N = ln); add b2
#pragma unroll
  for (int t = 0; t < 12; ++t) {
    int c = (nbase + t) * 16 + ln;
    float b2v = b2[c];
#pragma unroll
    for (int rr = 0; rr < 8; ++rr) {
      int R = bm * 64 + wm * 16 + rr + 8 * g;
      out[(size_t)R * NC + c] = acc[t][rr] + b2v;
    }
  }
}

extern "C" void kernel_launch(void* const* d_in, const int* in_sizes, int n_in,
                              void* d_out, int out_size, void* d_ws, size_t ws_size,
                              hipStream_t stream) {
  const float* f0   = (const float*)d_in[0];
  const int*   seg  = (const int*)d_in[1];
  // d_in[2] = n_words scalar (== 128), constants baked in
  const float* w1   = (const float*)d_in[3];
  const float* b1   = (const float*)d_in[4];
  const float* gm   = (const float*)d_in[5];
  const float* bt   = (const float*)d_in[6];
  const float* w2   = (const float*)d_in[7];
  const float* b2   = (const float*)d_in[8];
  float* out = (float*)d_out;

  char* ws = (char*)d_ws;
  size_t off = 0;
  auto carve = [&](size_t bytes) {
    char* p = ws + off;
    off += (bytes + 255) & ~(size_t)255;
    return p;
  };
  float*  f1  = (float*)carve((size_t)B_ * F1ROWS * D_ * 4);   // 1.6 MB
  float*  G   = (float*)carve((size_t)MPAD * D_ * 4);          // 1.6 MB
  float*  HG  = (float*)carve((size_t)MPAD * HID * 4);         // 4.3 MB
  half_t* w2h = (half_t*)carve((size_t)NC * HID * 2);          // 3.0 MB

  k_f1    <<<B_ * F1ROWS, 256, 0, stream>>>(f0, seg, f1);
  k_gather<<<MPAD,        256, 0, stream>>>(f1, G);
  k_w2h   <<<(NC * HID) / 1024, 256, 0, stream>>>(w2, w2h);
  k_hg    <<<MPAD,        256, 0, stream>>>(G, w1, HG);
  k_main  <<<(B_ * NW * GROWS) / 64, 512, 64 * HID * 2, stream>>>(
      HG, b1, gm, bt, w2h, b2, out);
}
